// dMaSIFConv_63419487093390
// MI455X (gfx1250) — compile-verified, matrix operand order checked by reading, not run on hardware
//
#include <hip/hip_runtime.h>
#include <math.h>

typedef __attribute__((ext_vector_type(16))) _Float16 v16h;
typedef __attribute__((ext_vector_type(8)))  float    v8f;

#define NPTS 2048
#define IDIM 16
#define HDIM 8
#define ODIM 16
#define NCUT 8

__device__ __forceinline__ float leaky(float x){ return x > 0.f ? x : 0.2f * x; }

__device__ __forceinline__ v16h zero16h(){
    v16h v;
#pragma unroll
    for (int k = 0; k < 16; ++k) v[k] = (_Float16)0.f;
    return v;
}
__device__ __forceinline__ v8f zero8f(){
    v8f v;
#pragma unroll
    for (int k = 0; k < 8; ++k) v[k] = 0.f;
    return v;
}

// Wave-local ordering for same-wave LDS produce->consume. CDNA5 LDS ops from
// one wave are serviced in order; we only need to stop compiler reordering.
__device__ __forceinline__ void wave_lds_fence(){
    __builtin_amdgcn_fence(__ATOMIC_ACQ_REL, "wavefront");
    __builtin_amdgcn_wave_barrier();
}

// ---------------------------------------------------------------------------
// Kernel 1: net_in MLP (16->8->8, leaky) + GroupNorm(groups=4 over 8 ch).
// Single block, 1024 threads, 2 points each; LDS-atomic reduction for stats.
// ---------------------------------------------------------------------------
__global__ __launch_bounds__(1024) void k_net_in(
    const float* __restrict__ feat,
    const float* __restrict__ Wi1, const float* __restrict__ bi1,
    const float* __restrict__ Wi2, const float* __restrict__ bi2,
    const float* __restrict__ gw,  const float* __restrict__ gb,
    float* __restrict__ out)           // N x 8
{
    __shared__ float s_sum[4], s_sq[4];
    const int t = threadIdx.x;
    if (t < 4){ s_sum[t] = 0.f; s_sq[t] = 0.f; }
    __syncthreads();

    float h2[2][HDIM];
    float lsum[4] = {0,0,0,0}, lsq[4] = {0,0,0,0};
#pragma unroll
    for (int p = 0; p < 2; ++p){
        const int i = t + p * 1024;
        float h1[HDIM];
#pragma unroll
        for (int o = 0; o < HDIM; ++o){
            float s = bi1[o];
#pragma unroll
            for (int k = 0; k < IDIM; ++k) s += feat[i*IDIM + k] * Wi1[o*IDIM + k];
            h1[o] = leaky(s);
        }
#pragma unroll
        for (int o = 0; o < HDIM; ++o){
            float s = bi2[o];
#pragma unroll
            for (int k = 0; k < HDIM; ++k) s += h1[k] * Wi2[o*HDIM + k];
            const float v = leaky(s);
            h2[p][o] = v;
            lsum[o >> 1] += v; lsq[o >> 1] += v * v;
        }
    }
#pragma unroll
    for (int g = 0; g < 4; ++g){ atomicAdd(&s_sum[g], lsum[g]); atomicAdd(&s_sq[g], lsq[g]); }
    __syncthreads();

    const float inv_cnt = 1.0f / (2.0f * NPTS);   // 2 channels/group * N
#pragma unroll
    for (int p = 0; p < 2; ++p){
        const int i = t + p * 1024;
#pragma unroll
        for (int c = 0; c < HDIM; ++c){
            const int g = c >> 1;
            const float m = s_sum[g] * inv_cnt;
            const float v = s_sq[g] * inv_cnt - m * m;
            const float r = rsqrtf(v + 1e-5f);
            out[i*HDIM + c] = (h2[p][c] - m) * r * gw[c] + gb[c];
        }
    }
}

// ---------------------------------------------------------------------------
// Kernel 2: all-pairs windowed conv. One block per i (128 thr = 4 waves),
// each wave independently owns 32 16-j tiles. Two V_WMMA_F32_16X16X32_F16
// per tile implement the (16x3)@(3x8) and (16x8)@(8x8) conv GEMMs with f32
// accumulation; window/exp and the weighted j-reduction stay fp32 VALU.
// No block barriers in the hot loop: each wave has a private LDS region and
// same-wave LDS ordering is guaranteed by hardware (+ compiler fence).
// ---------------------------------------------------------------------------
__global__ __launch_bounds__(128) void k_pair(
    const float* __restrict__ pts, const float* __restrict__ nuv,
    const float* __restrict__ fin,
    const float* __restrict__ Wc1, const float* __restrict__ bc1,
    const float* __restrict__ Wc2, const float* __restrict__ bc2,
    float* __restrict__ fout)          // N x 8
{
    __shared__ _Float16 s_x1h[4][128];  // per-wave 16x8 f16 transpose buffer
    __shared__ float    s_f  [4][128];  // per-wave staged f-tile (16 j x 8 h)
    __shared__ float    s_win[4][16];   // per-wave window values
    __shared__ float    s_acc[HDIM];

    const int i    = blockIdx.x;
    const int tid  = threadIdx.x;
    const int wid  = tid >> 5;
    const int lane = tid & 31;
    const int n    = lane & 15;
    const bool hi  = lane >= 16;

    if (tid < HDIM) s_acc[tid] = 0.f;

    const float SC = 0.0785674201f;    // 1/(sqrt(2)*RADIUS)
    const float pix = pts[i*3+0]*SC, piy = pts[i*3+1]*SC, piz = pts[i*3+2]*SC;
    float nv[9];
#pragma unroll
    for (int a = 0; a < 9; ++a) nv[a] = nuv[i*9 + a];

    // B operands, hoisted: columns striped across lanes 0-15; K on low half only.
    v16h b1 = zero16h();               // (Wc1^T) padded 3->32 in K
    v16h b2 = zero16h();               // (Wc2^T) padded 8->32 in K
    if (!hi && n < NCUT){
#pragma unroll
        for (int k = 0; k < 3; ++k) b1[k] = (_Float16)Wc1[n*3 + k];
#pragma unroll
        for (int k = 0; k < NCUT; ++k) b2[k] = (_Float16)Wc2[n*NCUT + k];
    }
    const float bias1 = bc1[n & 7];
    const float bias2 = bc2[n & 7];
    const v8f c0 = zero8f();

    float acc = 0.f;
    __syncthreads();                    // covers s_acc init only

    for (int tile = wid; tile < NPTS/16; tile += 4){
        const int j0 = tile * 16;
        const int j  = j0 + n;          // both lane-halves cover all 16 j's

        // Prefetch this wave's next tile (speculative; dropped if OOB).
        __builtin_prefetch(pts + (j0 + 64)*3, 0, 3);
        __builtin_prefetch(fin + (j0 + 64)*HDIM, 0, 3);

        const float dx = pts[j*3+0]*SC - pix;
        const float dy = pts[j*3+1]*SC - piy;
        const float dz = pts[j*3+2]*SC - piz;
        const float X0 = nv[0]*dx + nv[1]*dy + nv[2]*dz;
        const float X1 = nv[3]*dx + nv[4]*dy + nv[5]*dz;
        const float X2 = nv[6]*dx + nv[7]*dy + nv[8]*dz;
        const float ndot = nv[0]*nuv[j*9+0] + nv[1]*nuv[j*9+1] + nv[2]*nuv[j*9+2];
        const float sc2  = 2.f - ndot;
        const float w    = __expf(-(dx*dx + dy*dy + dz*dz) * sc2 * sc2);

        // Stage window + f-tile (lanes 0-15, coalesced 2x b128 per lane).
        if (!hi){
            s_win[wid][n] = w;
            const float4* fj = (const float4*)(fin + j*HDIM);
            *(float4*)&s_f[wid][n*8 + 0] = fj[0];
            *(float4*)&s_f[wid][n*8 + 4] = fj[1];
        }

        // GEMM1: A = local coords (16 j-rows x K=3, zero-padded to K=32)
        v16h a1 = zero16h();
        if (!hi){ a1[0] = (_Float16)X0; a1[1] = (_Float16)X1; a1[2] = (_Float16)X2; }
        v8f d1 = __builtin_amdgcn_wmma_f32_16x16x32_f16(false, a1, false, b1,
                                                        (short)0, c0, false, false);

        // Transpose X1 through LDS as f16 (D: VGPR k = row m, lane = column).
        if (n < NCUT){
            const int mrow = hi ? 8 : 0;
#pragma unroll
            for (int k = 0; k < 8; ++k)
                s_x1h[wid][(k + mrow)*8 + n] = (_Float16)fmaxf(d1[k] + bias1, 0.f);
        }
        wave_lds_fence();

        // GEMM2: reload X1 rows as A operand (one 16B LDS read per low lane).
        v16h a2 = zero16h();
        if (!hi){
            union { uint4 u; _Float16 h[8]; } t;
            t.u = *(const uint4*)&s_x1h[wid][n*8];
#pragma unroll
            for (int k = 0; k < NCUT; ++k) a2[k] = t.h[k];
        }
        v8f d2 = __builtin_amdgcn_wmma_f32_16x16x32_f16(false, a2, false, b2,
                                                        (short)0, c0, false, false);

        // lane n (valid n<8) holds X2[j0+k+off][h=n] in VGPR k
        const int off = hi ? 8 : 0;
        const int hm  = n & 7;
#pragma unroll
        for (int k = 0; k < 8; ++k){
            const float x2 = fmaxf(d2[k] + bias2, 0.f);
            acc += s_win[wid][k + off] * x2 * s_f[wid][(k + off)*8 + hm];
        }
        wave_lds_fence();               // next iter's stores must not pass reads
    }

    // Combine the two lane-halves (j 0-7 set + j 8-15 set) per h, then block.
    acc += __shfl_xor(acc, 16, 32);
    if (lane < HDIM) atomicAdd(&s_acc[lane], acc);
    __syncthreads();
    if (tid < HDIM) fout[i*HDIM + tid] = s_acc[tid];
}

// ---------------------------------------------------------------------------
// Kernel 3: net_out MLP (8->16->16, leaky) + GroupNorm(groups=4 over 16 ch).
// ---------------------------------------------------------------------------
__global__ __launch_bounds__(1024) void k_net_out(
    const float* __restrict__ fin,
    const float* __restrict__ Wo1, const float* __restrict__ bo1,
    const float* __restrict__ Wo2, const float* __restrict__ bo2,
    const float* __restrict__ gw,  const float* __restrict__ gb,
    float* __restrict__ out)           // N x 16
{
    __shared__ float s_sum[4], s_sq[4];
    const int t = threadIdx.x;
    if (t < 4){ s_sum[t] = 0.f; s_sq[t] = 0.f; }
    __syncthreads();

    float o2[2][ODIM];
    float lsum[4] = {0,0,0,0}, lsq[4] = {0,0,0,0};
#pragma unroll
    for (int p = 0; p < 2; ++p){
        const int i = t + p * 1024;
        float f[HDIM];
#pragma unroll
        for (int k = 0; k < HDIM; ++k) f[k] = fin[i*HDIM + k];
        float o1[ODIM];
#pragma unroll
        for (int o = 0; o < ODIM; ++o){
            float s = bo1[o];
#pragma unroll
            for (int k = 0; k < HDIM; ++k) s += f[k] * Wo1[o*HDIM + k];
            o1[o] = leaky(s);
        }
#pragma unroll
        for (int o = 0; o < ODIM; ++o){
            float s = bo2[o];
#pragma unroll
            for (int k = 0; k < ODIM; ++k) s += o1[k] * Wo2[o*ODIM + k];
            const float v = leaky(s);
            o2[p][o] = v;
            lsum[o >> 2] += v; lsq[o >> 2] += v * v;
        }
    }
#pragma unroll
    for (int g = 0; g < 4; ++g){ atomicAdd(&s_sum[g], lsum[g]); atomicAdd(&s_sq[g], lsq[g]); }
    __syncthreads();

    const float inv_cnt = 1.0f / (4.0f * NPTS);   // 4 channels/group * N
#pragma unroll
    for (int p = 0; p < 2; ++p){
        const int i = t + p * 1024;
#pragma unroll
        for (int c = 0; c < ODIM; ++c){
            const int g = c >> 2;
            const float m = s_sum[g] * inv_cnt;
            const float v = s_sq[g] * inv_cnt - m * m;
            const float r = rsqrtf(v + 1e-5f);
            out[i*ODIM + c] = (o2[p][c] - m) * r * gw[c] + gb[c];
        }
    }
}

extern "C" void kernel_launch(void* const* d_in, const int* in_sizes, int n_in,
                              void* d_out, int out_size, void* d_ws, size_t ws_size,
                              hipStream_t stream)
{
    (void)in_sizes; (void)n_in; (void)out_size; (void)ws_size;
    const float* points = (const float*)d_in[0];
    const float* nuv    = (const float*)d_in[1];
    const float* feat   = (const float*)d_in[2];
    const float* Wi1    = (const float*)d_in[3];
    const float* bi1    = (const float*)d_in[4];
    const float* Wi2    = (const float*)d_in[5];
    const float* bi2    = (const float*)d_in[6];
    const float* gin_w  = (const float*)d_in[7];
    const float* gin_b  = (const float*)d_in[8];
    const float* Wc1    = (const float*)d_in[9];
    const float* bc1    = (const float*)d_in[10];
    const float* Wc2    = (const float*)d_in[11];
    const float* bc2    = (const float*)d_in[12];
    const float* Wo1    = (const float*)d_in[13];
    const float* bo1    = (const float*)d_in[14];
    const float* Wo2    = (const float*)d_in[15];
    const float* bo2    = (const float*)d_in[16];
    const float* gout_w = (const float*)d_in[17];
    const float* gout_b = (const float*)d_in[18];

    float* out   = (float*)d_out;
    float* ws_f  = (float*)d_ws;              // N*8 floats
    float* ws_f2 = ws_f + NPTS * HDIM;        // N*8 floats

    k_net_in <<<1, 1024, 0, stream>>>(feat, Wi1, bi1, Wi2, bi2, gin_w, gin_b, ws_f);
    k_pair   <<<NPTS, 128, 0, stream>>>(points, nuv, ws_f, Wc1, bc1, Wc2, bc2, ws_f2);
    k_net_out<<<1, 1024, 0, stream>>>(ws_f2, Wo1, bo1, Wo2, bo2, gout_w, gout_b, out);
}
